// PureLNN_16234976379200
// MI455X (gfx1250) — matrix operand order repaired and checked by reading, not compiled, not run on hardware
//
#include <hip/hip_runtime.h>

// ---------------------------------------------------------------------------
// Liquid-NN forward for MI455X (gfx1250, wave32, WMMA).
//
//   a1   = relu(x @ W1^T + b1)                  (once)
//   s1_t = a1 * (1 - d1^t)                      (closed form of the s1 scan)
//   s2_t = d2*s2_{t-1} + (1-d2)*relu(s1_t @ W2^T + b2)
//   out  = s2_T @ W3^T + b3
//
// The per-step scale c1_t = 1-d1^t lies on the K axis of the step GEMM, so it
// is applied to the *resident A-fragments* with v_pk_mul_bf16 (32 packed ops
// per wave per step).  W2 sits read-only in LDS in B-fragment layout; the
// step loop needs a single barrier (double-buffered 128-entry c1 vector).
// ---------------------------------------------------------------------------

typedef __bf16 bf16;
typedef __attribute__((ext_vector_type(16))) __bf16 v16bf;
typedef __attribute__((ext_vector_type(8)))  __bf16 v8bf;
typedef __attribute__((ext_vector_type(8)))  float  v8f;

#define BATCH   65536
#define IN_DIM  784
#define HID     128
#define OUT_DIM 10
#define TSTEPS  10
#define BM      128      // batch rows per workgroup
#define KCHUNKS 13       // ceil(784/64), zero-padded to 832

// LDS byte offsets (total 135680 B; 2 WGs still fit in a 320 KB WGP)
#define OFF_A1   0        // [128][128] bf16 a1 (later reused to stage s2)
#define OFF_W2   32768    // [128][128] bf16 W2 row-major (== B-frag layout)
#define OFF_XW   65536    // 2 x { xs[128][64], ws[128][64] } ping-pong
#define OFF_W3   131072   // [16][128] bf16 W3 zero-padded to N=16
#define OFF_C1   135168   // 2 x [128] bf16 c1_t double buffer
#define SMEM_BYTES 135680

__device__ __forceinline__ v8f wmma_bf16(v16bf a, v16bf b, v8f c) {
  return __builtin_amdgcn_wmma_f32_16x16x32_bf16(false, a, false, b,
                                                 (short)0, c, false, false);
}

// packed bf16 multiply (VOP3P V_PK_MUL_BF16), 8 ops per 16-element fragment
__device__ __forceinline__ v16bf pkmul_bf16(v16bf a, v16bf b) {
  union U { v16bf h; unsigned int u[8]; } A, B, D;
  A.h = a; B.h = b;
#pragma unroll
  for (int i = 0; i < 8; ++i) {
    unsigned int x = A.u[i], y = B.u[i], z;
    asm("v_pk_mul_bf16 %0, %1, %2" : "=v"(z) : "v"(x), "v"(y));
    D.u[i] = z;
  }
  return D.h;
}

__device__ __forceinline__ float decay_from_logtau(float lt) {
  float tau = expf(lt);
  tau = fminf(fmaxf(tau, 0.1f), 100.0f);
  return expf(-1.0f / tau);
}

// A-fragment (bf16 16x32, ISA 7.12.2): lane L<16 -> row m=L, K={0..7,16..23};
// lanes 16..31 -> row m=L-16, K={8..15,24..31}  (per 32-wide K chunk kc).
__device__ __forceinline__ v16bf load_afrag(const bf16* base, int rowstride,
                                            int m0, int kc, int lane) {
  const int l16 = lane & 15;
  const int kb  = kc * 32 + ((lane >> 4) ? 8 : 0);
  const bf16* p = base + (m0 + l16) * rowstride + kb;
  v8bf lo = *(const v8bf*)p;
  v8bf hh = *(const v8bf*)(p + 16);
  v16bf r;
#pragma unroll
  for (int i = 0; i < 8; ++i) { r[i] = lo[i]; r[i + 8] = hh[i]; }
  return r;
}

// same gather pattern against a single 128-entry vector (broadcast per lane half)
__device__ __forceinline__ v16bf load_rowfrag(const bf16* row, int kc, int lane) {
  const int kb = kc * 32 + ((lane >> 4) ? 8 : 0);
  v8bf lo = *(const v8bf*)(row + kb);
  v8bf hh = *(const v8bf*)(row + kb + 16);
  v16bf r;
#pragma unroll
  for (int i = 0; i < 8; ++i) { r[i] = lo[i]; r[i + 8] = hh[i]; }
  return r;
}

// B-fragment (bf16 32x16): lane L<16 -> column n=L, K={0..7,16..23};
// lanes 16..31 -> K={8..15,24..31}.  LDS [n][k] row-major == W row-major.
__device__ __forceinline__ v16bf load_bfrag(const bf16* base, int rowstride,
                                            int n0, int kc, int lane) {
  const int l16 = lane & 15;
  const int kb  = kc * 32 + ((lane >> 4) ? 8 : 0);
  const bf16* p = base + (n0 + l16) * rowstride + kb;
  v8bf lo = *(const v8bf*)p;
  v8bf hh = *(const v8bf*)(p + 16);
  v16bf r;
#pragma unroll
  for (int i = 0; i < 8; ++i) { r[i] = lo[i]; r[i + 8] = hh[i]; }
  return r;
}

__global__ void __launch_bounds__(256, 1)
lnn_fused_kernel(const float* __restrict__ x,  const float* __restrict__ W1,
                 const float* __restrict__ b1, const float* __restrict__ W2,
                 const float* __restrict__ b2, const float* __restrict__ W3,
                 const float* __restrict__ b3, const float* __restrict__ lt1,
                 const float* __restrict__ lt2, float* __restrict__ out)
{
  extern __shared__ unsigned char smem[];
  bf16* a1s = (bf16*)(smem + OFF_A1);
  bf16* W2b = (bf16*)(smem + OFF_W2);
  bf16* W3b = (bf16*)(smem + OFF_W3);
  bf16* c1s = (bf16*)(smem + OFF_C1);

  const int tid  = threadIdx.x;
  const int lane = tid & 31;
  const int wv   = tid >> 5;          // wave 0..7
  const int l16  = lane & 15;
  const int hi8  = (lane >> 4) ? 8 : 0;
  const int m0   = wv * 16;           // this wave's 16-row strip
  const int r0   = blockIdx.x * BM;   // batch row base

  // ---- per-lane column parameters (n = nt*16 + l16) ----
  float b1n[8], b2n[8], d2n[8], od2n[8];
#pragma unroll
  for (int nt = 0; nt < 8; ++nt) {
    b1n[nt] = b1[nt * 16 + l16];
    b2n[nt] = b2[nt * 16 + l16];
    d2n[nt] = decay_from_logtau(lt2[nt * 16 + l16]);
    od2n[nt] = 1.0f - d2n[nt];
  }
  const float b3n = (l16 < OUT_DIM) ? b3[l16] : 0.0f;

  // c1 producer: thread j (<128) owns hidden unit j, keeps d1^t in f32
  const float d1j = (tid < HID) ? decay_from_logtau(lt1[tid]) : 0.0f;
  float p1 = 1.0f;

  // ---- phase 0: park W2 (bf16, row-major) and padded W3 in LDS ----
  {
    const int i = tid >> 1, jb = (tid & 1) * 64;
    const float* wr = W2 + i * HID + jb;
    bf16* dst = W2b + i * HID + jb;
#pragma unroll 16
    for (int e = 0; e < 64; ++e) dst[e] = (bf16)wr[e];
  }
#pragma unroll
  for (int e = 0; e < 8; ++e) {
    int idx = tid * 8 + e;
    int n = idx >> 7, k = idx & 127;
    W3b[idx] = (bf16)((n < OUT_DIM) ? W3[n * HID + k] : 0.0f);
  }

  // ---- phase 1: a1 = relu(x @ W1^T + b1), K in 64-wide ping-pong chunks ----
  v8f acc[8];
#pragma unroll
  for (int nt = 0; nt < 8; ++nt)
#pragma unroll
    for (int v = 0; v < 8; ++v) acc[nt][v] = 0.0f;

  const int lr = tid >> 1;            // row this thread stages (0..127)
  const int lh = (tid & 1) * 32;      // low/high 32 of the 64-wide chunk
  const float* xrow = x  + (size_t)(r0 + lr) * IN_DIM;
  const float* wrow = W1 + (size_t)lr * IN_DIM;

  for (int kc = 0; kc < KCHUNKS; ++kc) {
    bf16* xsb = (bf16*)(smem + OFF_XW + (kc & 1) * 32768);
    bf16* wsb = xsb + 128 * 64;
    const int kb = kc * 64 + lh;
    bf16* xd = xsb + lr * 64 + lh;
    bf16* wd = wsb + lr * 64 + lh;
    if (kb + 32 <= IN_DIM) {
#pragma unroll
      for (int e = 0; e < 32; ++e) xd[e] = (bf16)xrow[kb + e];
#pragma unroll
      for (int e = 0; e < 32; ++e) wd[e] = (bf16)wrow[kb + e];
    } else {                          // tail (784 = 12*64 + 16), zero-pad
#pragma unroll
      for (int e = 0; e < 32; ++e) xd[e] = (bf16)((kb + e < IN_DIM) ? xrow[kb + e] : 0.0f);
#pragma unroll
      for (int e = 0; e < 32; ++e) wd[e] = (bf16)((kb + e < IN_DIM) ? wrow[kb + e] : 0.0f);
    }
    if (kc + 1 < KCHUNKS) {
      __builtin_prefetch(xrow + kb + 64, 0, 0);   // global_prefetch_b8
      __builtin_prefetch(wrow + kb + 64, 0, 0);
    }
    __syncthreads();   // single barrier: ping-pong covers WAR across iterations
#pragma unroll
    for (int kcc = 0; kcc < 2; ++kcc) {
      v16bf af = load_afrag(xsb, 64, m0, kcc, lane);
#pragma unroll
      for (int nt = 0; nt < 8; ++nt) {
        v16bf bfr = load_bfrag(wsb, 64, nt * 16, kcc, lane);
        acc[nt] = wmma_bf16(af, bfr, acc[nt]);
      }
    }
  }

  // bias + relu; stash a1 tile in LDS in [m][k] (A-fragment friendly) layout
#pragma unroll
  for (int nt = 0; nt < 8; ++nt)
#pragma unroll
    for (int v = 0; v < 8; ++v) {
      float a = fmaxf(acc[nt][v] + b1n[nt], 0.0f);
      a1s[(m0 + v + hi8) * HID + nt * 16 + l16] = (bf16)a;
    }
  __syncthreads();

  // a1 A-fragments stay resident in VGPRs for the whole recurrence
  v16bf afr[4];
#pragma unroll
  for (int kc2 = 0; kc2 < 4; ++kc2) afr[kc2] = load_afrag(a1s, HID, m0, kc2, lane);

  v8f s2[8];
#pragma unroll
  for (int nt = 0; nt < 8; ++nt)
#pragma unroll
    for (int v = 0; v < 8; ++v) s2[nt][v] = 0.0f;

  // ---- phase 2: T steps; scale resident A-frags by c1_t, W2 read-only ----
  for (int step = 0; step < TSTEPS; ++step) {
    if (tid < HID) {
      p1 *= d1j;                                   // d1^t kept in f32
      c1s[(step & 1) * HID + tid] = (bf16)(1.0f - p1);
    }
    __syncthreads();                               // only barrier per step
    const bf16* cbuf = c1s + (step & 1) * HID;

    v8f r[8];
#pragma unroll
    for (int nt = 0; nt < 8; ++nt)
#pragma unroll
      for (int v = 0; v < 8; ++v) r[nt][v] = b2n[nt];   // start from bias

#pragma unroll
    for (int kc2 = 0; kc2 < 4; ++kc2) {
      v16bf c1a = load_rowfrag(cbuf, kc2, lane);
      v16bf as_ = pkmul_bf16(afr[kc2], c1a);       // s1_t fragment
#pragma unroll
      for (int nt = 0; nt < 8; ++nt) {
        v16bf bfr = load_bfrag(W2b, HID, nt * 16, kc2, lane);
        r[nt] = wmma_bf16(as_, bfr, r[nt]);
      }
    }
#pragma unroll
    for (int nt = 0; nt < 8; ++nt)
#pragma unroll
      for (int v = 0; v < 8; ++v)
        s2[nt][v] = d2n[nt] * s2[nt][v] + od2n[nt] * fmaxf(r[nt][v], 0.0f);
  }

  // ---- phase 3: out = s2 @ W3^T + b3 via WMMA (N padded 10->16) ----
#pragma unroll
  for (int nt = 0; nt < 8; ++nt)      // stage s2 into a1s (region reused)
#pragma unroll
    for (int v = 0; v < 8; ++v)
      a1s[(m0 + v + hi8) * HID + nt * 16 + l16] = (bf16)s2[nt][v];
  __syncthreads();

  v8f o;
#pragma unroll
  for (int v = 0; v < 8; ++v) o[v] = b3n;
#pragma unroll
  for (int kc2 = 0; kc2 < 4; ++kc2) {
    v16bf af2 = load_afrag(a1s, HID, m0, kc2, lane);
    v16bf bf3 = load_bfrag(W3b, HID, 0, kc2, lane);
    o = wmma_bf16(af2, bf3, o);
  }
  if (l16 < OUT_DIM) {
#pragma unroll
    for (int v = 0; v < 8; ++v)
      out[(size_t)(r0 + m0 + v + hi8) * OUT_DIM + l16] = o[v];
  }
}

extern "C" void kernel_launch(void* const* d_in, const int* in_sizes, int n_in,
                              void* d_out, int out_size, void* d_ws, size_t ws_size,
                              hipStream_t stream) {
  const float* x   = (const float*)d_in[0];
  const float* W1  = (const float*)d_in[1];
  const float* b1  = (const float*)d_in[2];
  const float* W2  = (const float*)d_in[3];
  const float* b2  = (const float*)d_in[4];
  const float* W3  = (const float*)d_in[5];
  const float* b3  = (const float*)d_in[6];
  const float* lt1 = (const float*)d_in[7];
  const float* lt2 = (const float*)d_in[8];
  float* out = (float*)d_out;

  dim3 grid(BATCH / BM);              // 512 workgroups
  dim3 block(256);                    // 8 wave32 waves
  hipLaunchKernelGGL(lnn_fused_kernel, grid, block, SMEM_BYTES, stream,
                     x, W1, b1, W2, b2, W3, b3, lt1, lt2, out);
}
